// LSTMgate_77781857730587
// MI455X (gfx1250) — matrix operand order repaired and checked by reading, not compiled, not run on hardware
//
#include <hip/hip_runtime.h>

// ---------------------------------------------------------------------------
// LSTM recurrent net on MI455X (gfx1250), wave32 + WMMA bf16.
//
// B=4096, T=256, NOISE=64, H=512, IN=68 (padded to 96), 4H=2048.
// One workgroup (256 threads = 8 waves) owns a 16-row batch tile and runs all
// 256 recurrent steps locally (the recurrence is independent per batch row).
// All matmuls are M=16 WMMA (v_wmma_f32_16x16x32_bf16), f32 accumulate.
// Weights converted fp32->bf16 once into workspace (~3 MB of d_ws); they stay
// resident in the 192 MB L2 for the whole recurrence. h double-buffered bf16
// in LDS, c f32 in LDS. Loops rolled to avoid spills; activations use the
// CDNA5 hardware tanh (V_TANH_F32) when available.
// ---------------------------------------------------------------------------

#define B_     4096
#define T_     256
#define NOISE_ 64
#define H_     512
#define GD_    2
#define IN_    68
#define INP_   96        // IN padded to multiple of 32 for K-chunks
#define G4_    2048      // 4*H
#define DPR_   24.0f
#define BT_    16        // batch rows per workgroup

typedef __attribute__((ext_vector_type(16))) __bf16 v16bf;
typedef __attribute__((ext_vector_type(8)))  float  v8f;

union AB16 { v16bf v; uint4 q[2]; };

// Workspace layout (bytes)
#define WS_W1B   0                         // 512*512*2   = 524288
#define WS_WHHB  (512 * 1024)              // 2048*512*2  = 2097152
#define WS_WIHB  (512 * 1024 + 2097152)    // 2048*96*2   = 393216
// total = 3014656 bytes (assumes ws_size >= ~3 MB)

// Hardware tanh (V_TANH_F32) if the toolchain exposes it; libm fallback.
#if defined(__has_builtin)
#  if __has_builtin(__builtin_amdgcn_tanhf)
#    define TANHF(x) __builtin_amdgcn_tanhf(x)
#  endif
#endif
#ifndef TANHF
#  define TANHF(x) tanhf(x)
#endif

// sigmoid(x) = 0.5 * tanh(x/2) + 0.5  (exact identity -> one TRANS op)
__device__ inline float sigmoidf_(float x) { return fmaf(0.5f, TANHF(0.5f * x), 0.5f); }

// 16x32 bf16 A-tile from row-major LDS (row stride `stride` elems), WMMA A
// layout (ISA 7.12.2): lanes 0-15 row M=lane hold K {0..7,16..23}, lanes
// 16-31 hold K {8..15,24..31}.  Two ds_load_b128 per lane.
__device__ inline v16bf load_a_tile(const __bf16* base, int stride, int kbase, int lane) {
    const int m   = lane & 15;
    const int sel = lane >> 4;
    AB16 u;
    u.q[0] = *reinterpret_cast<const uint4*>(base + m * stride + kbase + sel * 8);
    u.q[1] = *reinterpret_cast<const uint4*>(base + m * stride + kbase + 16 + sel * 8);
    return u.v;
}

// 32x16 bf16 B-tile from row-major W (N x K, ldk elems): B[k][n]=W[n][k].
// Lanes 0-15: column n=lane, K {0..7,16..23}; lanes 16-31: K {8..15,24..31}.
// Two global_load_b128 per lane.
__device__ inline v16bf load_b_tile(const __bf16* W, int ldk, int nbase, int kbase, int lane) {
    const int n    = lane & 15;
    const int koff = (lane >> 4) * 8;
    const uint4* p = reinterpret_cast<const uint4*>(W + (size_t)(nbase + n) * ldk + kbase + koff);
    AB16 u;
    u.q[0] = p[0];
    u.q[1] = p[2];
    return u.v;
}

__device__ inline v8f wmma_bf16(v16bf a, v16bf b, v8f c) {
    return __builtin_amdgcn_wmma_f32_16x16x32_bf16(
        /*neg_a=*/false, a, /*neg_b=*/false, b,
        /*c_mod=*/(short)0, c, /*reuse_a=*/false, /*reuse_b=*/false);
}

// ---------------------------------------------------------------------------
// Kernel 0: convert weights fp32 -> bf16 (W_ih padded 68 -> 96 cols with 0).
// ---------------------------------------------------------------------------
__global__ __launch_bounds__(256)
void cvt_weights_kernel(const float* __restrict__ W1, const float* __restrict__ Whh,
                        const float* __restrict__ Wih,
                        unsigned short* __restrict__ W1b_u, unsigned short* __restrict__ Whhb_u,
                        unsigned short* __restrict__ Wihb_u) {
    __bf16* W1b  = reinterpret_cast<__bf16*>(W1b_u);
    __bf16* Whhb = reinterpret_cast<__bf16*>(Whhb_u);
    __bf16* Wihb = reinterpret_cast<__bf16*>(Wihb_u);
    const int idx = blockIdx.x * blockDim.x + threadIdx.x;
    if (idx < H_ * H_)   W1b[idx]  = (__bf16)W1[idx];
    if (idx < G4_ * H_)  Whhb[idx] = (__bf16)Whh[idx];
    if (idx < G4_ * INP_) {
        const int r = idx / INP_, c = idx % INP_;
        const float v = (c < IN_) ? Wih[r * IN_ + c] : 0.0f;
        Wihb[idx] = (__bf16)v;
    }
}

// ---------------------------------------------------------------------------
// Kernel 1: persistent per-batch-tile recurrence, all T steps in one launch.
// ---------------------------------------------------------------------------
__global__ __launch_bounds__(256)
void lstm_recurrent_kernel(const float* __restrict__ noise,     // (B,T,64)
                           const float* __restrict__ cluster,   // (B,1,1)
                           const float* __restrict__ gap,       // (B,T+1,2)
                           const float* __restrict__ b_ih,      // (2048)
                           const float* __restrict__ b_hh,      // (2048)
                           const float* __restrict__ b1,        // (512)
                           const float* __restrict__ W2,        // (1,512)
                           const float* __restrict__ b2,        // (1)
                           const unsigned short* __restrict__ W1b_u,   // (512,512) bf16
                           const unsigned short* __restrict__ Whhb_u,  // (2048,512) bf16
                           const unsigned short* __restrict__ Wihb_u,  // (2048,96) bf16
                           float* __restrict__ out)             // (B,T+1,3)
{
    const __bf16* W1b  = reinterpret_cast<const __bf16*>(W1b_u);
    const __bf16* Whhb = reinterpret_cast<const __bf16*>(Whhb_u);
    const __bf16* Wihb = reinterpret_cast<const __bf16*>(Wihb_u);

    __shared__ __bf16 h_lds[2][BT_ * H_];   // hidden state, double buffered (32 KB)
    __shared__ float  c_lds[BT_ * H_];      // cell state f32 (32 KB)
    __shared__ __bf16 z_lds[BT_ * H_];      // out_net hidden (16 KB)
    __shared__ __bf16 x_lds[BT_ * INP_];    // padded LSTM input (3 KB)
    __shared__ float  red[BT_ * 16];        // dp reduction scratch
    __shared__ float  dp_sh[BT_];

    const int tid  = threadIdx.x;
    const int lane = tid & 31;
    const int wave = tid >> 5;              // 0..7
    const int b0   = blockIdx.x * BT_;

    // h0 = 0, c0 = 0
    for (int i = tid; i < BT_ * H_; i += 256) {
        h_lds[0][i] = (__bf16)0.0f;
        c_lds[i]    = 0.0f;
    }
    __syncthreads();

    int cur = 0;
    const float b2s = b2[0];

    for (int t = 0; t <= T_; ++t) {
        const __bf16* hcur = h_lds[cur];

        // ---------------- phase 1: z = tanh(h @ W1^T + b1) -----------------
        // 4 N-tile accumulators per wave share each A-tile load.
        {
            v8f acc[4];
            #pragma unroll
            for (int i = 0; i < 4; ++i) {
                const float bias = b1[(wave + 8 * i) * 16 + (lane & 15)];
                #pragma unroll
                for (int r = 0; r < 8; ++r) acc[i][r] = bias;
            }
            #pragma clang loop unroll_count(2)
            for (int kc = 0; kc < 16; ++kc) {
                v16bf a = load_a_tile(hcur, H_, kc * 32, lane);
                #pragma unroll
                for (int i = 0; i < 4; ++i)
                    acc[i] = wmma_bf16(a, load_b_tile(W1b, H_, (wave + 8 * i) * 16, kc * 32, lane),
                                       acc[i]);
            }
            const int n = lane & 15, mtop = (lane >> 4) * 8;
            #pragma unroll
            for (int i = 0; i < 4; ++i) {
                const int nbase = (wave + 8 * i) * 16;
                #pragma unroll
                for (int r = 0; r < 8; ++r)
                    z_lds[(mtop + r) * H_ + nbase + n] = (__bf16)TANHF(acc[i][r]);
            }
        }
        __syncthreads();

        // ---------------- phase 2: dp = tanh(z @ W2^T + b2) * 24 -----------
        {
            const int m = tid >> 4;   // 0..15
            const int j = tid & 15;
            float s = 0.0f;
            for (int k = j; k < H_; k += 16)
                s += (float)z_lds[m * H_ + k] * W2[k];
            red[m * 16 + j] = s;
        }
        __syncthreads();
        if ((tid & 15) == 0) {
            const int m = tid >> 4;
            float s = 0.0f;
            #pragma unroll
            for (int j = 0; j < 16; ++j) s += red[m * 16 + j];
            const float dp = TANHF(s + b2s) * DPR_;
            dp_sh[m] = dp;
            const size_t grow = ((size_t)(b0 + m) * (T_ + 1) + t);
            out[grow * 3 + 0] = gap[grow * GD_ + 0];
            out[grow * 3 + 1] = gap[grow * GD_ + 1];
            out[grow * 3 + 2] = dp;
        }
        __syncthreads();

        if (t == T_) break;   // final step only emits x_last

        // ---------------- phase 3: build lstm_in (bf16, padded to 96) ------
        {
            const int m = tid >> 4;   // 16 threads per batch row
            const int j = tid & 15;
            const size_t grow = ((size_t)(b0 + m) * (T_ + 1) + t);
            #pragma clang loop unroll(disable)
            for (int col = j; col < INP_; col += 16) {
                float v;
                if (col < GD_)             v = gap[grow * GD_ + col];
                else if (col == 2)         v = dp_sh[m];
                else if (col < 3 + NOISE_) v = noise[(((size_t)(b0 + m)) * T_ + t) * NOISE_ + (col - 3)];
                else if (col == IN_ - 1)   v = cluster[b0 + m];
                else                       v = 0.0f;
                x_lds[m * INP_ + col] = (__bf16)v;
            }
        }
        __syncthreads();

        // ---------------- phase 4: gates + LSTM cell -----------------------
        __bf16* hnext = h_lds[cur ^ 1];
        #pragma clang loop unroll(disable)
        for (int i = 0; i < 4; ++i) {
            const int nb  = (wave + 8 * i) * 16;      // hidden-column tile base
            const int col = nb + (lane & 15);
            const float bi = b_ih[col]          + b_hh[col];
            const float bf = b_ih[H_ + col]     + b_hh[H_ + col];
            const float bg = b_ih[2 * H_ + col] + b_hh[2 * H_ + col];
            const float bo = b_ih[3 * H_ + col] + b_hh[3 * H_ + col];
            v8f gi, gf, gg, go;
            #pragma unroll
            for (int r = 0; r < 8; ++r) { gi[r] = bi; gf[r] = bf; gg[r] = bg; go[r] = bo; }

            // input part: K = 96 (3 chunks), A from x_lds, B from W_ih (bf16)
            #pragma clang loop unroll(disable)
            for (int kc = 0; kc < 3; ++kc) {
                v16bf a = load_a_tile(x_lds, INP_, kc * 32, lane);
                gi = wmma_bf16(a, load_b_tile(Wihb, INP_, nb,          kc * 32, lane), gi);
                gf = wmma_bf16(a, load_b_tile(Wihb, INP_, H_ + nb,     kc * 32, lane), gf);
                gg = wmma_bf16(a, load_b_tile(Wihb, INP_, 2 * H_ + nb, kc * 32, lane), gg);
                go = wmma_bf16(a, load_b_tile(Wihb, INP_, 3 * H_ + nb, kc * 32, lane), go);
            }
            // hidden part: K = 512 (16 chunks), A from h, B from W_hh (bf16)
            #pragma clang loop unroll_count(2)
            for (int kc = 0; kc < 16; ++kc) {
                v16bf a = load_a_tile(hcur, H_, kc * 32, lane);
                gi = wmma_bf16(a, load_b_tile(Whhb, H_, nb,          kc * 32, lane), gi);
                gf = wmma_bf16(a, load_b_tile(Whhb, H_, H_ + nb,     kc * 32, lane), gf);
                gg = wmma_bf16(a, load_b_tile(Whhb, H_, 2 * H_ + nb, kc * 32, lane), gg);
                go = wmma_bf16(a, load_b_tile(Whhb, H_, 3 * H_ + nb, kc * 32, lane), go);
            }
            // elementwise LSTM cell; lane owns (m = (lane>>4)*8 + r, col)
            #pragma unroll
            for (int r = 0; r < 8; ++r) {
                const int m = (lane >> 4) * 8 + r;
                const float iv = sigmoidf_(gi[r]);
                const float fv = sigmoidf_(gf[r]);
                const float gv = TANHF(gg[r]);
                const float ov = sigmoidf_(go[r]);
                const float c2 = fv * c_lds[m * H_ + col] + iv * gv;
                c_lds[m * H_ + col] = c2;
                hnext[m * H_ + col] = (__bf16)(ov * TANHF(c2));
            }
        }
        cur ^= 1;
        __syncthreads();
    }
}

// ---------------------------------------------------------------------------
extern "C" void kernel_launch(void* const* d_in, const int* in_sizes, int n_in,
                              void* d_out, int out_size, void* d_ws, size_t ws_size,
                              hipStream_t stream) {
    const float* noise   = (const float*)d_in[0];
    const float* cluster = (const float*)d_in[1];
    const float* gap     = (const float*)d_in[2];
    const float* W_ih    = (const float*)d_in[3];
    const float* W_hh    = (const float*)d_in[4];
    const float* b_ih    = (const float*)d_in[5];
    const float* b_hh    = (const float*)d_in[6];
    const float* W1      = (const float*)d_in[7];
    const float* b1      = (const float*)d_in[8];
    const float* W2      = (const float*)d_in[9];
    const float* b2      = (const float*)d_in[10];
    float* out = (float*)d_out;

    char* ws = (char*)d_ws;
    unsigned short* W1b  = (unsigned short*)(ws + WS_W1B);
    unsigned short* Whhb = (unsigned short*)(ws + WS_WHHB);
    unsigned short* Wihb = (unsigned short*)(ws + WS_WIHB);

    const int total = G4_ * H_;  // largest conversion job
    cvt_weights_kernel<<<(total + 255) / 256, 256, 0, stream>>>(W1, W_hh, W_ih, W1b, Whhb, Wihb);

    lstm_recurrent_kernel<<<B_ / BT_, 256, 0, stream>>>(
        noise, cluster, gap, b_ih, b_hh, b1, W2, b2, W1b, Whhb, Wihb, out);
}
